// AsyncChunkTriangleMultiplicationOutgoing_858993459583
// MI455X (gfx1250) — matrix operand order repaired and checked by reading, not compiled, not run on hardware
//
#include <hip/hip_runtime.h>

typedef __attribute__((ext_vector_type(16))) __bf16 v16bf;
typedef __attribute__((ext_vector_type(8)))  __bf16 v8bf;
typedef __attribute__((ext_vector_type(8)))  float  v8f;
typedef __attribute__((ext_vector_type(4)))  float  v4f;

#define DEV __device__ __forceinline__

constexpr int N  = 512;   // sequence length
constexpr int D  = 128;   // pair channel dim
constexpr float LN_EPS = 1e-5f;

// ---------------------------------------------------------------------------
// WMMA helper: D = A(16x32 bf16) x B(32x16 bf16) + C(16x16 f32)
// ---------------------------------------------------------------------------
DEV v8f wmma_bf16(v16bf a, v16bf b, v8f c) {
  return __builtin_amdgcn_wmma_f32_16x16x32_bf16(false, a, false, b, (short)0, c,
                                                 false, false);
}

DEV float sigmoidf(float x) { return 1.0f / (1.0f + __expf(-x)); }

// A fragment (16x32, MxK) from a row-major [m][k] bf16 source, row stride ldr.
// ISA layout: lane = h*16+m holds K = h*8+{0..7} (elems 0-7) and 16+h*8+{0..7}.
DEV v16bf load_a(const __bf16* row0, int ldr, int k0, int h, int m) {
  const __bf16* rp = row0 + m * ldr + k0 + h * 8;
  v8bf lo = *(const v8bf*)(rp);
  v8bf hi = *(const v8bf*)(rp + 16);
  v16bf r;
#pragma unroll
  for (int i = 0; i < 8; ++i) { r[i] = lo[i]; r[i + 8] = hi[i]; }
  return r;
}

// B fragment (32x16, KxN) where B[k][n] = W[n0+n][k], W row-major stride ldw.
// ISA layout: lane = h*16+n holds column n, K = h*8+{0..7} (elems 0-7) and 16+h*8+{0..7}.
DEV v16bf load_b(const __bf16* W, int ldw, int n0, int k0, int h, int n) {
  const __bf16* rp = W + (n0 + n) * ldw + k0 + h * 8;
  v8bf lo = *(const v8bf*)(rp);
  v8bf hi = *(const v8bf*)(rp + 16);
  v16bf r;
#pragma unroll
  for (int i = 0; i < 8; ++i) { r[i] = lo[i]; r[i + 8] = hi[i]; }
  return r;
}

// ---------------------------------------------------------------------------
// Kernel 0: f32 -> bf16 weight cast
// ---------------------------------------------------------------------------
__global__ void k_cvt_bf16(const float* __restrict__ s, __bf16* __restrict__ d,
                           int n) {
  int i = blockIdx.x * blockDim.x + threadIdx.x;
  if (i < n) d[i] = (__bf16)s[i];
}

// ---------------------------------------------------------------------------
// Kernel 1: LN1 + lr/gate/og projections, fused mask+sigmoid epilogue.
// Block = 256 threads (8 waves), 128 positions. Wave w owns rows 16w..16w+15.
// Outputs: leftT/rightT bf16 in channel-major [x][c][y] (k contiguous),
//          gbuf bf16 in [x][c][y].
// ---------------------------------------------------------------------------
__global__ __launch_bounds__(256)
void k_pre(const float* __restrict__ Zraw, const float* __restrict__ maskrow,
           const float* __restrict__ g1, const float* __restrict__ b1,
           const float* __restrict__ b_lr, const float* __restrict__ b_gate,
           const float* __restrict__ b_og,
           const __bf16* __restrict__ Wlr, const __bf16* __restrict__ Wg,
           const __bf16* __restrict__ Wog,
           __bf16* __restrict__ leftT, __bf16* __restrict__ rightT,
           __bf16* __restrict__ gbuf) {
  __shared__ __bf16 lnbuf[128][136];   // 136 = 128 + pad (bank-conflict free)
  const int tid  = threadIdx.x;
  const int wave = tid >> 5;
  const int lane = tid & 31;
  const int p0   = blockIdx.x * 128;

  // ---- Phase 1: LayerNorm (each wave: 16 rows; lane covers 4 columns) ----
  {
    v4f gg = *(const v4f*)(g1 + lane * 4);
    v4f bb = *(const v4f*)(b1 + lane * 4);
    for (int rr = 0; rr < 16; ++rr) {
      int r = wave * 16 + rr;
      v4f xv = *(const v4f*)(Zraw + (size_t)(p0 + r) * D + lane * 4);
      float s  = xv.x + xv.y + xv.z + xv.w;
      float sq = xv.x * xv.x + xv.y * xv.y + xv.z * xv.z + xv.w * xv.w;
#pragma unroll
      for (int off = 16; off; off >>= 1) {
        s  += __shfl_xor(s, off, 32);
        sq += __shfl_xor(sq, off, 32);
      }
      float mu  = s * (1.0f / 128.0f);
      float var = sq * (1.0f / 128.0f) - mu * mu;
      float rs  = rsqrtf(var + LN_EPS);
#pragma unroll
      for (int t = 0; t < 4; ++t)
        lnbuf[r][lane * 4 + t] = (__bf16)((xv[t] - mu) * rs * gg[t] + bb[t]);
    }
  }
  __syncthreads();

  // ---- Phase 2: WMMA projections ----
  const int h  = lane >> 4, hm = lane & 15;
  const int rowbase = p0 + wave * 16;
  const int x = rowbase >> 9, yb = rowbase & (N - 1);

  v16bf A[4];
#pragma unroll
  for (int ks = 0; ks < 4; ++ks)
    A[ks] = load_a(&lnbuf[wave * 16][0], 136, ks * 32, h, hm);

  float maskv[8];
#pragma unroll
  for (int v = 0; v < 8; ++v) maskv[v] = maskrow[rowbase + 8 * h + v];

  // lr projection (256 outputs) gated elementwise by sigmoid(gate).
  // Double-buffered B fragments: next k-step's loads issue before the WMMAs.
  for (int t = 0; t < 16; ++t) {
    int e0 = t * 16, e = e0 + hm;
    v8f aL = {}, aG = {};
    v16bf bL = load_b(Wlr, D, e0, 0, h, hm);
    v16bf bG = load_b(Wg,  D, e0, 0, h, hm);
#pragma unroll
    for (int ks = 0; ks < 4; ++ks) {
      // final-iteration loads are dead (DCE'd); addresses stay inside ws
      v16bf nbL = load_b(Wlr, D, e0, ks * 32 + 32, h, hm);
      v16bf nbG = load_b(Wg,  D, e0, ks * 32 + 32, h, hm);
      aL = wmma_bf16(A[ks], bL, aL);
      aG = wmma_bf16(A[ks], bG, aG);
      bL = nbL; bG = nbG;
    }
    float blr = b_lr[e], bg = b_gate[e];
    v8bf outp;
#pragma unroll
    for (int v = 0; v < 8; ++v) {
      float lr = (aL[v] + blr) * maskv[v];
      outp[v] = (__bf16)(lr * sigmoidf(aG[v] + bg));
    }
    __bf16* dst = (e < 128)
        ? (leftT  + (size_t)(x * 128 + e)        * N + yb + 8 * h)
        : (rightT + (size_t)(x * 128 + (e - 128)) * N + yb + 8 * h);
    *(v8bf*)dst = outp;
  }

  // output gate: g = sigmoid(Z @ W_og^T + b_og)
  for (int t = 0; t < 8; ++t) {
    int e0 = t * 16, e = e0 + hm;
    v8f aO = {};
    v16bf bO = load_b(Wog, D, e0, 0, h, hm);
#pragma unroll
    for (int ks = 0; ks < 4; ++ks) {
      v16bf nbO = load_b(Wog, D, e0, ks * 32 + 32, h, hm);
      aO = wmma_bf16(A[ks], bO, aO);
      bO = nbO;
    }
    float bo = b_og[e];
    v8bf og;
#pragma unroll
    for (int v = 0; v < 8; ++v) og[v] = (__bf16)sigmoidf(aO[v] + bo);
    *(v8bf*)(gbuf + (size_t)(x * 128 + e) * N + yb + 8 * h) = og;
  }
}

// ---------------------------------------------------------------------------
// Kernel 2: triangle core. p[i,j,c] = sum_k left[i,k,c]*right[j,k,c]
// Block = 16x16 (i,j) tile x all 128 channels; wave owns 16 channels.
// Two channels interleaved per iteration (two independent WMMA chains) and
// double-buffered fragments (loads lead the WMMAs by one k-step).
// ---------------------------------------------------------------------------
__global__ __launch_bounds__(256)
void k_tri(const __bf16* __restrict__ leftT, const __bf16* __restrict__ rightT,
           float* __restrict__ P) {
  const int tid  = threadIdx.x;
  const int wave = tid >> 5, lane = tid & 31;
  const int h = lane >> 4, hm = lane & 15;
  const int i0 = (blockIdx.x >> 5) * 16;
  const int j0 = (blockIdx.x & 31) * 16;
  constexpr int LDC = 128 * N;  // stride between spatial rows (channel-major)

  for (int cc = 0; cc < 16; cc += 2) {
    const int c0 = wave * 16 + cc;
    const __bf16* A0r = leftT  + ((size_t)i0 * 128 + c0) * N;
    const __bf16* B0r = rightT + ((size_t)j0 * 128 + c0) * N;
    const __bf16* A1r = A0r + N;   // channel c0+1
    const __bf16* B1r = B0r + N;

    v8f acc0 = {}, acc1 = {};
    v16bf a0 = load_a(A0r, LDC, 0, h, hm);
    v16bf b0 = load_b(B0r, LDC, 0, 0, h, hm);
    v16bf a1 = load_a(A1r, LDC, 0, h, hm);
    v16bf b1 = load_b(B1r, LDC, 0, 0, h, hm);
#pragma unroll
    for (int ks = 0; ks < 16; ++ks) {
      const int kn = ks * 32 + 32;
      __builtin_prefetch(A0r + hm * LDC + kn + 96, 0, 1);
      __builtin_prefetch(B0r + hm * LDC + kn + 96, 0, 1);
      // next k-step fragments (dead + DCE'd on the last unrolled iteration;
      // addresses remain inside the contiguous ws arena)
      v16bf na0 = load_a(A0r, LDC, kn, h, hm);
      v16bf nb0 = load_b(B0r, LDC, 0, kn, h, hm);
      v16bf na1 = load_a(A1r, LDC, kn, h, hm);
      v16bf nb1 = load_b(B1r, LDC, 0, kn, h, hm);
      acc0 = wmma_bf16(a0, b0, acc0);
      acc1 = wmma_bf16(a1, b1, acc1);
      a0 = na0; b0 = nb0; a1 = na1; b1 = nb1;
    }
#pragma unroll
    for (int v = 0; v < 8; ++v) {
      size_t row = (size_t)(i0 + 8 * h + v) * N + (j0 + hm);
      P[row * 128 + c0]     = acc0[v];
      P[row * 128 + c0 + 1] = acc1[v];
    }
  }
}

// ---------------------------------------------------------------------------
// Kernel 3: LN2(p) @ W_out + fused gate * residual epilogue.
// Same block structure as k_pre (128 positions / block).
// ---------------------------------------------------------------------------
__global__ __launch_bounds__(256)
void k_post(const float* __restrict__ P, const float* __restrict__ g2,
            const float* __restrict__ b2, const __bf16* __restrict__ Wout,
            const float* __restrict__ obias, const __bf16* __restrict__ gbuf,
            const float* __restrict__ Zraw, float* __restrict__ out) {
  __shared__ __bf16 lnbuf[128][136];
  const int tid  = threadIdx.x;
  const int wave = tid >> 5, lane = tid & 31;
  const int p0   = blockIdx.x * 128;

  {
    v4f gg = *(const v4f*)(g2 + lane * 4);
    v4f bb = *(const v4f*)(b2 + lane * 4);
    for (int rr = 0; rr < 16; ++rr) {
      int r = wave * 16 + rr;
      v4f xv = *(const v4f*)(P + (size_t)(p0 + r) * 128 + lane * 4);
      float s  = xv.x + xv.y + xv.z + xv.w;
      float sq = xv.x * xv.x + xv.y * xv.y + xv.z * xv.z + xv.w * xv.w;
#pragma unroll
      for (int off = 16; off; off >>= 1) {
        s  += __shfl_xor(s, off, 32);
        sq += __shfl_xor(sq, off, 32);
      }
      float mu  = s * (1.0f / 128.0f);
      float var = sq * (1.0f / 128.0f) - mu * mu;
      float rs  = rsqrtf(var + LN_EPS);
#pragma unroll
      for (int t = 0; t < 4; ++t)
        lnbuf[r][lane * 4 + t] = (__bf16)((xv[t] - mu) * rs * gg[t] + bb[t]);
    }
  }
  __syncthreads();

  const int h = lane >> 4, hm = lane & 15;
  const int rowbase = p0 + wave * 16;
  const int x = rowbase >> 9, yb = rowbase & (N - 1);

  v16bf A[4];
#pragma unroll
  for (int ks = 0; ks < 4; ++ks)
    A[ks] = load_a(&lnbuf[wave * 16][0], 136, ks * 32, h, hm);

  for (int t = 0; t < 8; ++t) {
    int e0 = t * 16, e = e0 + hm;
    v8f acc = {};
    v16bf bO = load_b(Wout, 128, e0, 0, h, hm);
#pragma unroll
    for (int ks = 0; ks < 4; ++ks) {
      v16bf nbO = load_b(Wout, 128, e0, ks * 32 + 32, h, hm);
      acc = wmma_bf16(A[ks], bO, acc);
      bO = nbO;
    }
    float ob = obias[e];
    v8bf gv = *(const v8bf*)(gbuf + (size_t)(x * 128 + e) * N + yb + 8 * h);
#pragma unroll
    for (int v = 0; v < 8; ++v) {
      size_t idx = (size_t)(rowbase + 8 * h + v) * 128 + e;
      out[idx] = Zraw[idx] + (float)gv[v] * (acc[v] + ob);
    }
  }
}

// ---------------------------------------------------------------------------
extern "C" void kernel_launch(void* const* d_in, const int* in_sizes, int n_in,
                              void* d_out, int out_size, void* d_ws,
                              size_t ws_size, hipStream_t stream) {
  (void)in_sizes; (void)n_in; (void)out_size; (void)ws_size;
  const float* Zraw  = (const float*)d_in[0];
  const float* maskr = (const float*)d_in[1];
  const float* ln1g  = (const float*)d_in[2];
  const float* ln1b  = (const float*)d_in[3];
  const float* Wlr   = (const float*)d_in[4];
  const float* blr   = (const float*)d_in[5];
  const float* Wgate = (const float*)d_in[6];
  const float* bgate = (const float*)d_in[7];
  const float* Wog   = (const float*)d_in[8];
  const float* bog   = (const float*)d_in[9];
  const float* ln2g  = (const float*)d_in[10];
  const float* ln2b  = (const float*)d_in[11];
  const float* Wout  = (const float*)d_in[12];
  const float* obias = (const float*)d_in[13];
  float* out = (float*)d_out;

  char* ws = (char*)d_ws;
  size_t off = 0;
  __bf16* wWlr  = (__bf16*)(ws + off); off += (size_t)256 * 128 * 2;
  __bf16* wWg   = (__bf16*)(ws + off); off += (size_t)256 * 128 * 2;
  __bf16* wWog  = (__bf16*)(ws + off); off += (size_t)128 * 128 * 2;
  __bf16* wWout = (__bf16*)(ws + off); off += (size_t)128 * 128 * 2;
  __bf16* leftT  = (__bf16*)(ws + off); off += (size_t)128 * N * N * 2;
  __bf16* rightT = (__bf16*)(ws + off); off += (size_t)128 * N * N * 2;
  __bf16* gbuf   = (__bf16*)(ws + off); off += (size_t)128 * N * N * 2;
  float*  P      = (float*)(ws + off);  off += (size_t)N * N * 128 * 4;

  k_cvt_bf16<<<128, 256, 0, stream>>>(Wlr,   wWlr,  256 * 128);
  k_cvt_bf16<<<128, 256, 0, stream>>>(Wgate, wWg,   256 * 128);
  k_cvt_bf16<<<64,  256, 0, stream>>>(Wog,   wWog,  128 * 128);
  k_cvt_bf16<<<64,  256, 0, stream>>>(Wout,  wWout, 128 * 128);

  k_pre<<<(N * N) / 128, 256, 0, stream>>>(Zraw, maskr, ln1g, ln1b, blr, bgate,
                                           bog, wWlr, wWg, wWog, leftT, rightT,
                                           gbuf);
  k_tri<<<(N / 16) * (N / 16), 256, 0, stream>>>(leftT, rightT, P);
  k_post<<<(N * N) / 128, 256, 0, stream>>>(P, ln2g, ln2b, wWout, obias, gbuf,
                                            Zraw, out);
}
